// Attention_59004260712651
// MI455X (gfx1250) — matrix operand order repaired and checked by reading, not compiled, not run on hardware
//
#include <hip/hip_runtime.h>
#include <hip/hip_bf16.h>
#include <math.h>

// ---------------------------------------------------------------------------
// Shapes (fixed by the reference): B=256, T=5, S=64, F=256, H=4, Fh=64
// ---------------------------------------------------------------------------
#define B_DIM 256
#define T_DIM 5
#define S_DIM 64
#define F_DIM 256
#define FH_DIM 64
#define HB_DIM 1024                 // H*B
#define MROWS (B_DIM * T_DIM * S_DIM)   // 81920 rows for the F x F projections
#define NTOT (MROWS * F_DIM)            // 20,971,520 elements of x / q / v / o

typedef __attribute__((ext_vector_type(16))) __bf16 v16bf;
typedef __attribute__((ext_vector_type(8)))  float  v8f;

union FragBF { v16bf v; unsigned int u[8]; };

// round-to-nearest-even fp32 -> bf16
__device__ __forceinline__ __bf16 f2bf(float f) {
  union { float f; unsigned int u; } c; c.f = f;
  unsigned int r = c.u + 0x7FFFu + ((c.u >> 16) & 1u);
  union { unsigned short s; __bf16 b; } o; o.s = (unsigned short)(r >> 16);
  return o.b;
}

__device__ __forceinline__ float sigmoidf(float x) {
  return 1.0f / (1.0f + __expf(-x));
}

// LDS byte offset of a __shared__ object: generic pointer low 32 bits are the
// LDS-aperture offset (ISA §10.2).
__device__ __forceinline__ unsigned lds_off(const void* p) {
  return (unsigned)(size_t)p;
}

// ASYNCcnt-tracked 16-byte global -> LDS copy (GV mode: vdst = LDS byte
// offset, vaddr = 64-bit global address). CDNA5 ISA.
__device__ __forceinline__ void async_g2l_b128(unsigned ldsByteOff, const void* gaddr) {
  asm volatile("global_load_async_to_lds_b128 %0, %1, off"
               :
               : "v"(ldsByteOff), "v"((unsigned long long)(size_t)gaddr)
               : "memory");
}

__device__ __forceinline__ void wait_asynccnt0() {
  asm volatile("s_wait_asynccnt 0x0" ::: "memory");
}

// --- WMMA fragment builders (layouts per CDNA5 ISA §7.12.2) -----
// A fragment: 16x32 bf16, row-major LDS tile, rowStride in bf16 elements.
// lane l (l&15)=M;  l<16: K = {0..7,16..23}+kbase ; l>=16: K = {8..15,24..31}+kbase
__device__ __forceinline__ v16bf load_a_frag(const __bf16* tile, int rowStride,
                                             int row0, int kbase, int lane) {
  int m    = lane & 15;
  int koff = (lane < 16) ? 0 : 8;
  const unsigned int* p = (const unsigned int*)(tile + (size_t)(row0 + m) * rowStride);
  FragBF f;
  int b0 = (kbase + koff) >> 1;
  int b1 = (kbase + 16 + koff) >> 1;
#pragma unroll
  for (int i = 0; i < 4; ++i) { f.u[i] = p[b0 + i]; f.u[4 + i] = p[b1 + i]; }
  return f.v;
}

// B fragment: 32x16 bf16 from a K-contiguous ("transposed") LDS tile wt[n][k].
// lane l: N=(l&15); l<16 -> K={0..7,16..23}+kbase ; l>=16 -> K={8..15,24..31}+kbase
__device__ __forceinline__ v16bf load_b_frag(const __bf16* wt, int kStride,
                                             int n0, int kbase, int lane) {
  int n  = n0 + (lane & 15);
  int kb = kbase + ((lane < 16) ? 0 : 8);
  const unsigned int* p = (const unsigned int*)(wt + (size_t)n * kStride);
  FragBF f;
  int b0 = kb >> 1;
#pragma unroll
  for (int i = 0; i < 4; ++i) { f.u[i] = p[b0 + i]; f.u[4 + i] = p[b0 + 8 + i]; }
  return f.v;
}

// ---------------------------------------------------------------------------
// K1: k[b,s,f] = sum_t u_t[t] * x[b,t,s,f]          (B*S*F = 4,194,304 elems)
// ---------------------------------------------------------------------------
__global__ __launch_bounds__(256) void k_proj_kernel(const float* __restrict__ x,
                                                     const float* __restrict__ u_t,
                                                     float* __restrict__ kbsf) {
  int g = blockIdx.x * 256 + threadIdx.x;       // (b*64+s)*256+f
  int b  = g >> 14;                              // / (S*F)
  int sf = g & 16383;
  float acc = 0.f;
#pragma unroll
  for (int t = 0; t < T_DIM; ++t)
    acc += u_t[t] * x[(size_t)(b * T_DIM + t) * 16384 + sf];
  kbsf[g] = acc;
}

// ---------------------------------------------------------------------------
// K1b: prior[t,s,c] = 1/(sqrt(2pi)*sigma[t,s]) * exp(-dis^2 / (2 sigma^2))
// ---------------------------------------------------------------------------
__global__ __launch_bounds__(256) void prior_kernel(const float* __restrict__ dis,
                                                    const float* __restrict__ sigma,
                                                    float* __restrict__ prior) {
  int g = blockIdx.x * 256 + threadIdx.x;        // t*4096 + s*64 + c
  if (g >= T_DIM * S_DIM * S_DIM) return;
  float sg = sigma[g >> 6];                      // index t*64+s  (sigma is [T,S,1])
  float d  = dis[g];
  prior[g] = (0.3989422804014327f / sg) * __expf(-d * d / (2.f * sg * sg));
}

// ---------------------------------------------------------------------------
// K2: fused q = x*Wq, v = x*Wv.  C tile 32 rows x 32 cols per block.
// 8 waves: waves 0-3 -> q tiles, waves 4-7 -> v tiles (2x2 tiles of 16x16).
// LDS: A 32x256 bf16 (16KB) + WqT 32x256 (16KB) + WvT 32x256 (16KB) = 48KB
// ---------------------------------------------------------------------------
__global__ __launch_bounds__(256) void qv_gemm_kernel(const float* __restrict__ x,
                                                      const float* __restrict__ Wq,
                                                      const float* __restrict__ Wv,
                                                      float* __restrict__ qout,
                                                      float* __restrict__ vout) {
  __shared__ __attribute__((aligned(16))) __bf16 aT [32 * 256];
  __shared__ __attribute__((aligned(16))) __bf16 wqT[32 * 256];   // [n][k]
  __shared__ __attribute__((aligned(16))) __bf16 wvT[32 * 256];   // [n][k]
  int bm = blockIdx.x >> 3;            // 2560 row blocks of 32
  int bn = blockIdx.x & 7;             // 8 col blocks of 32
  int rowBase = bm * 32, colBase = bn * 32;
  int tid = threadIdx.x;
  for (int i = tid; i < 32 * 256; i += 256) {
    int r = i >> 8, k = i & 255;
    aT[i] = f2bf(x[(size_t)(rowBase + r) * F_DIM + k]);
    int n = r;                          // reuse decomposition for the weight tiles
    wqT[i] = f2bf(Wq[(size_t)k * F_DIM + colBase + n]);
    wvT[i] = f2bf(Wv[(size_t)k * F_DIM + colBase + n]);
  }
  __syncthreads();
  int wave = tid >> 5, lane = tid & 31;
  int mat  = wave >> 2;                 // 0 -> q, 1 -> v
  int tile = wave & 3;
  int rt = (tile & 1) * 16;             // row tile in A block
  int ct = (tile >> 1) * 16;            // col tile in 32-col block
  const __bf16* wT = mat ? wvT : wqT;
  v8f acc = {};
#pragma unroll
  for (int kb = 0; kb < 256; kb += 32) {
    v16bf a = load_a_frag(aT, 256, rt, kb, lane);
    v16bf b = load_b_frag(wT, 256, ct, kb, lane);
    acc = __builtin_amdgcn_wmma_f32_16x16x32_bf16(false, a, false, b,
                                                  (short)0, acc, false, false);
  }
  float* out = mat ? vout : qout;
  int col = colBase + ct + (lane & 15);
  int row0 = rowBase + rt + ((lane < 16) ? 0 : 8);
#pragma unroll
  for (int j = 0; j < 8; ++j)
    out[(size_t)(row0 + j) * F_DIM + col] = acc[j];
}

// ---------------------------------------------------------------------------
// K3: per (n,t): score = sigmoid((q2[n,t] @ k2[n])/8) - bias[n%B,t] ; *prior ;
//     column-sum over s -> sums[n,t,c].  One 64x64x64 GEMM per block.
// q2 view of flat q: q2[n,t,s,fh] = q[((n*T+t)*S+s)*Fh + fh]
// k2[n,fh,c] = kbsf[b=n/4][c][f=(n%4)*64+fh]  (clean head split on F)
// Tiles staged fp32 via ASYNCcnt-tracked global_load_async_to_lds_b128
// (no VGPR round-trip), then converted to bf16 WMMA tiles in LDS.
// LDS: qF32 16KB + kF32 16KB + qT 8KB + kT 8KB + colSum = ~48.3KB
// ---------------------------------------------------------------------------
__global__ __launch_bounds__(256) void score_kernel(const float* __restrict__ q,
                                                    const float* __restrict__ kbsf,
                                                    const float* __restrict__ bias,
                                                    const float* __restrict__ prior,
                                                    float* __restrict__ sums) {
  __shared__ __attribute__((aligned(16))) float  qF32[64 * 64];
  __shared__ __attribute__((aligned(16))) float  kF32[64 * 64];   // [c][fh]
  __shared__ __attribute__((aligned(16))) __bf16 qT[64 * 64];
  __shared__ __attribute__((aligned(16))) __bf16 kT[64 * 64];     // [c][fh]
  __shared__ float colSum[64];
  int nb = blockIdx.x;                 // 0 .. HB*T-1
  int n  = nb / T_DIM, t = nb % T_DIM;
  int tid = threadIdx.x;
  int b = n >> 2, fbase = (n & 3) * FH_DIM, bb = n & (B_DIM - 1);

  // --- async global->LDS staging (fp32), 16-byte chunks per lane ----------
  const float* qbase = q + (size_t)nb * 4096;     // (n*T+t)*S*Fh, contiguous 16KB
  unsigned qLds = lds_off(qF32);
  unsigned kLds = lds_off(kF32);
#pragma unroll
  for (int j = 0; j < 4; ++j) {
    int cidx = tid + 256 * j;                     // 0..1023 chunks of 16B
    // q tile: fully contiguous
    async_g2l_b128(qLds + cidx * 16, (const char*)qbase + (size_t)cidx * 16);
    // k tile: row c = 64 contiguous floats at stride F_DIM
    int c = cidx >> 4, sub = cidx & 15;           // 16 chunks per 256B row
    const float* krow = kbsf + (size_t)(b * S_DIM + c) * F_DIM + fbase;
    async_g2l_b128(kLds + cidx * 16, (const char*)krow + sub * 16);
  }
  const float* biasB  = bias  + ((size_t)bb * T_DIM + t) * 4096;  // [s][c]
  const float* priorB = prior + (size_t)t * 4096;                 // [s][c]
  __builtin_prefetch(biasB + tid * 16, 0, 0);     // pull epilogue rows toward L2
  __builtin_prefetch(priorB + tid * 16, 0, 3);
  if (tid < 64) colSum[tid] = 0.f;
  wait_asynccnt0();
  __syncthreads();

  // --- convert staged fp32 -> bf16 WMMA tiles -----------------------------
  for (int i = tid; i < 4096; i += 256) {
    qT[i] = f2bf(qF32[i]);
    kT[i] = f2bf(kF32[i]);
  }
  __syncthreads();

  int wave = tid >> 5, lane = tid & 31;
#pragma unroll
  for (int tt = 0; tt < 2; ++tt) {
    int tile = wave * 2 + tt;           // 16 tiles of 16x16
    int st = (tile >> 2) * 16;          // s tile
    int ct = (tile & 3) * 16;           // c tile
    v8f acc = {};
    v16bf a0 = load_a_frag(qT, 64, st, 0, lane);
    v16bf b0 = load_b_frag(kT, 64, ct, 0, lane);
    acc = __builtin_amdgcn_wmma_f32_16x16x32_bf16(false, a0, false, b0,
                                                  (short)0, acc, false, false);
    v16bf a1 = load_a_frag(qT, 64, st, 32, lane);
    v16bf b1 = load_b_frag(kT, 64, ct, 32, lane);
    acc = __builtin_amdgcn_wmma_f32_16x16x32_bf16(false, a1, false, b1,
                                                  (short)0, acc, false, false);
    int c   = ct + (lane & 15);
    int s0  = st + ((lane < 16) ? 0 : 8);
    float partial = 0.f;
#pragma unroll
    for (int j = 0; j < 8; ++j) {
      int s = s0 + j;
      float sc = sigmoidf(acc[j] * 0.125f);             // /sqrt(Fh)=8
      partial += (sc - biasB[s * 64 + c]) * priorB[s * 64 + c];
    }
    atomicAdd(&colSum[c], partial);
  }
  __syncthreads();
  if (tid < 64) sums[(size_t)nb * 64 + tid] = colSum[tid];
}

// ---------------------------------------------------------------------------
// K5: o = (v * scale) @ Wo ; scale per element = sums[flat>>6]
// LDS: A 32x256 (16KB) + WoT 64x256 (32KB) = 48KB; 8 waves = 2x4 tiles.
// ---------------------------------------------------------------------------
__global__ __launch_bounds__(256) void o_gemm_kernel(const float* __restrict__ v,
                                                     const float* __restrict__ sums,
                                                     const float* __restrict__ Wo,
                                                     float* __restrict__ oout) {
  __shared__ __attribute__((aligned(16))) __bf16 aT [32 * 256];
  __shared__ __attribute__((aligned(16))) __bf16 woT[64 * 256];   // [n][k]
  int bm = blockIdx.x >> 2;            // 2560 row blocks of 32
  int bn = blockIdx.x & 3;             // 4 col blocks of 64
  int rowBase = bm * 32, colBase = bn * 64;
  int tid = threadIdx.x;
  for (int i = tid; i < 32 * 256; i += 256) {
    int r = i >> 8, k = i & 255;
    size_t g = (size_t)(rowBase + r) * F_DIM + k;
    aT[i] = f2bf(v[g] * sums[g >> 6]);
  }
  for (int i = tid; i < 64 * 256; i += 256) {
    int n = i >> 8, k = i & 255;
    woT[i] = f2bf(Wo[(size_t)k * F_DIM + colBase + n]);
  }
  __syncthreads();
  int wave = tid >> 5, lane = tid & 31;
  int rt = (wave & 1) * 16;            // 2 row tiles
  int ct = (wave >> 1) * 16;           // 4 col tiles
  v8f acc = {};
#pragma unroll
  for (int kb = 0; kb < 256; kb += 32) {
    v16bf a = load_a_frag(aT, 256, rt, kb, lane);
    v16bf b = load_b_frag(woT, 256, ct, kb, lane);
    acc = __builtin_amdgcn_wmma_f32_16x16x32_bf16(false, a, false, b,
                                                  (short)0, acc, false, false);
  }
  int col = colBase + ct + (lane & 15);
  int row0 = rowBase + rt + ((lane < 16) ? 0 : 8);
#pragma unroll
  for (int j = 0; j < 8; ++j)
    oout[(size_t)(row0 + j) * F_DIM + col] = acc[j];
}

// ---------------------------------------------------------------------------
// K6: SE block per batch b: pooled avg/max over (s,f) per t, two tiny MLPs,
//     se[b,t] = (1-w)*sig(mlp(avg)) + w*sig(mlp(max))
// ---------------------------------------------------------------------------
__global__ __launch_bounds__(256) void se_kernel(const float* __restrict__ o,
                                                 const float* __restrict__ fc1w,
                                                 const float* __restrict__ fc1b,
                                                 const float* __restrict__ fc2w,
                                                 const float* __restrict__ fc2b,
                                                 const float* __restrict__ biliw,
                                                 float* __restrict__ se) {
  __shared__ float reds[256];
  __shared__ float redm[256];
  __shared__ float avg[T_DIM], mx[T_DIM];
  int b = blockIdx.x, tid = threadIdx.x;
  for (int t = 0; t < T_DIM; ++t) {
    const float* p = o + (size_t)(b * T_DIM + t) * (S_DIM * F_DIM);
    float s = 0.f, m = -3.4e38f;
    for (int i = tid; i < S_DIM * F_DIM; i += 256) {
      float vv = p[i]; s += vv; m = fmaxf(m, vv);
    }
    reds[tid] = s; redm[tid] = m;
    __syncthreads();
    for (int off = 128; off > 0; off >>= 1) {
      if (tid < off) {
        reds[tid] += reds[tid + off];
        redm[tid] = fmaxf(redm[tid], redm[tid + off]);
      }
      __syncthreads();
    }
    if (tid == 0) { avg[t] = reds[0] * (1.0f / (S_DIM * F_DIM)); mx[t] = redm[0]; }
    __syncthreads();
  }
  if (tid == 0) {
    float h1[25], h2[25];
    for (int j = 0; j < 25; ++j) {
      float a1 = fc1b[j], a2 = fc1b[j];
      for (int i = 0; i < T_DIM; ++i) {
        a1 += avg[i] * fc1w[i * 25 + j];
        a2 += mx[i]  * fc1w[i * 25 + j];
      }
      h1[j] = fmaxf(a1, 0.f); h2[j] = fmaxf(a2, 0.f);
    }
    float w = biliw[0];
    for (int k = 0; k < T_DIM; ++k) {
      float a1 = fc2b[k], a2 = fc2b[k];
      for (int j = 0; j < 25; ++j) {
        a1 += h1[j] * fc2w[j * T_DIM + k];
        a2 += h2[j] * fc2w[j * T_DIM + k];
      }
      se[b * T_DIM + k] = (1.f - w) * sigmoidf(a1) + w * sigmoidf(a2);
    }
  }
}

// ---------------------------------------------------------------------------
// K7: out = o * se[b,t] + x
// ---------------------------------------------------------------------------
__global__ __launch_bounds__(256) void final_kernel(const float* __restrict__ o,
                                                    const float* __restrict__ se,
                                                    const float* __restrict__ x,
                                                    float* __restrict__ out) {
  size_t g = (size_t)blockIdx.x * 256 + threadIdx.x;
  int bt = (int)(g >> 14);            // (b*T+t)
  int b  = bt / T_DIM, t = bt % T_DIM;
  out[g] = o[g] * se[b * T_DIM + t] + x[g];
}

// ---------------------------------------------------------------------------
extern "C" void kernel_launch(void* const* d_in, const int* in_sizes, int n_in,
                              void* d_out, int out_size, void* d_ws, size_t ws_size,
                              hipStream_t stream) {
  // setup_inputs order:
  const float* x     = (const float*)d_in[0];   // [B,T,S,F]
  const float* bias  = (const float*)d_in[1];   // [B,T,S,S]
  const float* Wq    = (const float*)d_in[2];   // [F,F]
  const float* Wv    = (const float*)d_in[3];   // [F,F]
  const float* Wo    = (const float*)d_in[4];   // [F,F]
  const float* u_t   = (const float*)d_in[5];   // [T]
  const float* dis   = (const float*)d_in[6];   // [T,S,S]
  const float* sigma = (const float*)d_in[7];   // [T,S,1]
  const float* fc1w  = (const float*)d_in[8];   // [T, 25]
  const float* fc1b  = (const float*)d_in[9];   // [25]
  const float* fc2w  = (const float*)d_in[10];  // [25, T]
  const float* fc2b  = (const float*)d_in[11];  // [T]
  const float* biliw = (const float*)d_in[12];  // [1]
  float* out = (float*)d_out;

  // workspace layout (floats); o reuses the q region after the score pass
  float* ws     = (float*)d_ws;
  float* q_ws   = ws;                           // NTOT   (later: o)
  float* v_ws   = q_ws + (size_t)NTOT;          // NTOT
  float* k_ws   = v_ws + (size_t)NTOT;          // B*S*F = 4,194,304
  float* pri_ws = k_ws + (size_t)(B_DIM * S_DIM * F_DIM);   // 20480
  float* sum_ws = pri_ws + (size_t)(T_DIM * S_DIM * S_DIM); // HB*T*S = 327680
  float* se_ws  = sum_ws + (size_t)(HB_DIM * T_DIM * S_DIM); // B*T = 1280
  float* o_ws   = q_ws;                         // alias: q dead after score_kernel

  // 1) k = einsum('t,btsf->bsf')
  k_proj_kernel<<<(B_DIM * S_DIM * F_DIM) / 256, 256, 0, stream>>>(x, u_t, k_ws);
  // 2) Gaussian prior table
  prior_kernel<<<(T_DIM * S_DIM * S_DIM + 255) / 256, 256, 0, stream>>>(dis, sigma, pri_ws);
  // 3) q, v projections (bf16 WMMA)
  qv_gemm_kernel<<<(MROWS / 32) * 8, 256, 0, stream>>>(x, Wq, Wv, q_ws, v_ws);
  // 4) attention scores -> column sums (async LDS staging + bf16 WMMA)
  score_kernel<<<HB_DIM * T_DIM, 256, 0, stream>>>(q_ws, k_ws, bias, pri_ws, sum_ws);
  // 5) o = (v * scale) @ Wo (bf16 WMMA), writes over the q region
  o_gemm_kernel<<<(MROWS / 32) * 4, 256, 0, stream>>>(v_ws, sum_ws, Wo, o_ws);
  // 6) SE gates
  se_kernel<<<B_DIM, 256, 0, stream>>>(o_ws, fc1w, fc1b, fc2w, fc2b, biliw, se_ws);
  // 7) out = o * se + x
  final_kernel<<<NTOT / 256, 256, 0, stream>>>(o_ws, se_ws, x, out);

  (void)in_sizes; (void)n_in; (void)out_size; (void)ws_size;
}
